// Dcrnn_73212012527869
// MI455X (gfx1250) — matrix operand order repaired and checked by reading, not compiled, not run on hardware
//
#include <hip/hip_runtime.h>
#include <hip/hip_bf16.h>
#include <math.h>
#include <stdint.h>

typedef __attribute__((ext_vector_type(16))) _Float16 v16h;
typedef __attribute__((ext_vector_type(8)))  float    v8f;

#define F_IN   32
#define F_HID  64
#define PRE_C  128   // [z-preact (64) | h-preact (64)]
#define BFRAG_HALVES (3 * 8 * 32 * 16)   // 12288 halves = 24 KB

// ---------------- zero-init (deterministic; harness poisons ws) ----------------
__global__ void zero_f32_kernel(float* __restrict__ p, long n) {
  long i = (long)blockIdx.x * blockDim.x + threadIdx.x;
  long stride = (long)gridDim.x * blockDim.x;
  for (; i < n; i += stride) p[i] = 0.0f;
}

// ---------------- weighted degrees ----------------
__global__ void degree_kernel(const int* __restrict__ src, const int* __restrict__ dst,
                              const float* __restrict__ w,
                              float* __restrict__ degOut, float* __restrict__ degIn, int E) {
  int e = blockIdx.x * blockDim.x + threadIdx.x;
  if (e >= E) return;
  float we = w[e];
  atomicAdd(&degOut[src[e]], we);
  atomicAdd(&degIn[dst[e]], we);
}

__global__ void invert_kernel(float* __restrict__ degOut, float* __restrict__ degIn, int n) {
  int i = blockIdx.x * blockDim.x + threadIdx.x;
  if (i >= n) return;
  float o = degOut[i]; degOut[i] = (o > 0.0f) ? 1.0f / o : 0.0f;
  float d = degIn[i];  degIn[i]  = (d > 0.0f) ? 1.0f / d : 0.0f;
}

// ---------------- diffusion propagation: lane = feature (wave32), 8 edges/block ----------------
__global__ void propagate_kernel(const int* __restrict__ src, const int* __restrict__ dst,
                                 const float* __restrict__ w, const float* __restrict__ x,
                                 const float* __restrict__ degOutInv, const float* __restrict__ degInInv,
                                 float* __restrict__ Txo, float* __restrict__ Txi, int E) {
  int f = threadIdx.x;                             // 0..31 feature (one wave per edge)
  int e = blockIdx.x * blockDim.y + threadIdx.y;   // edge id (wave-uniform)
  if (e >= E) return;
  int s = src[e], d = dst[e];
  float we = w[e];
  float cf = we * degOutInv[s];                    // forward: D_O^{-1} at source
  float cb = we * degInInv[d];                     // backward: D_I^{-1} at dest
  float xs = x[(long)s * F_IN + f];
  float xd = x[(long)d * F_IN + f];
  atomicAdd(&Txo[(long)d * F_IN + f], cf * xs);    // segment_sum over dst
  atomicAdd(&Txi[(long)s * F_IN + f], cb * xd);    // segment_sum over src
}

// ---------------- pack effective weights into WMMA B-fragment layout (f16) ----------------
// Effective (H0 == 0 => only first 32 rows of cin matter; reset gate is dead):
//   kstep0: W[0,0][:32] + W[1,0][:32]   (acts on X)
//   kstep1: W[0,1][:32]                 (acts on Txo)
//   kstep2: W[1,1][:32]                 (acts on Txi)
// Bfrag layout: [kstep(3)][ntile(8)][lane(32)][16 halves]
//   lane<16 : N = tileCol+lane,    K = kstep*32 + 0..15
//   lane>=16: N = tileCol+lane-16, K = kstep*32 + 16..31
__global__ void pack_weights_kernel(const float* __restrict__ Wz, const float* __restrict__ Wh,
                                    _Float16* __restrict__ Bfrag) {
  int t = blockIdx.x * blockDim.x + threadIdx.x;
  if (t >= 96 * PRE_C) return;
  int kg  = t / PRE_C;        // 0..95 reduced-K index
  int col = t % PRE_C;        // 0..127 output column
  int gate = col >> 6;        // 0 = z, 1 = h
  int oc   = col & 63;
  int kstep = kg >> 5;        // 0..2
  int kk    = kg & 31;        // input feature
  const float* W = gate ? Wh : Wz;   // [2, 2, 96, 64] row-major
  float v;
  if (kstep == 0)      v = W[((0*2 + 0)*96 + kk)*64 + oc] + W[((1*2 + 0)*96 + kk)*64 + oc];
  else if (kstep == 1) v = W[((0*2 + 1)*96 + kk)*64 + oc];
  else                 v = W[((1*2 + 1)*96 + kk)*64 + oc];
  int ntile = col >> 4;
  int lane  = (col & 15) | ((kk >> 4) << 4);
  int j     = kk & 15;
  Bfrag[(((kstep * 8 + ntile) * 32) + lane) * 16 + j] = (_Float16)v;
}

// ---------------- WMMA GEMM: Pre[N,128] = [X|Txo|Txi] (N x 96, f16) @ Bfrag (96 x 128) ----------------
// One wave per 16-row tile; 8 waves / block. B fragments staged Global->LDS once per
// block via CDNA5 async-to-LDS DMA (ASYNCcnt), WMMAs then feed from LDS.
__global__ void __launch_bounds__(256) gemm_wmma_kernel(
    const float* __restrict__ X, const float* __restrict__ Txo, const float* __restrict__ Txi,
    const _Float16* __restrict__ Bfrag, float* __restrict__ Pre, int nTiles) {
  __shared__ __align__(16) _Float16 sB[BFRAG_HALVES];   // 24 KB

  int tid  = threadIdx.x;
  int wave = tid >> 5;
  int lane = tid & 31;

  // ---- stage B-fragment table into LDS: 1536 x b128 chunks, 6 per thread ----
  {
    uint32_t ldsBase = (uint32_t)(uintptr_t)&sB[0];     // low 32 bits = LDS byte address
    const char* gBase = (const char*)Bfrag;
#pragma unroll
    for (int i = 0; i < 6; ++i) {
      int chunk = tid + i * 256;                        // 16-byte chunk index
      uint32_t laddr = ldsBase + (uint32_t)chunk * 16u;
      const char* gaddr = gBase + (size_t)chunk * 16u;
      asm volatile("global_load_async_to_lds_b128 %0, %1, off"
                   :: "v"(laddr), "v"(gaddr) : "memory");
    }
    asm volatile("s_wait_asynccnt 0" ::: "memory");     // my async writes to LDS done
  }
  __syncthreads();                                      // all waves' staging visible

  int tile = blockIdx.x * 8 + wave;
  if (tile >= nTiles) return;                 // wave-uniform: EXEC stays all-ones for WMMA

  // A-fragment (contiguous k order, as in Bfrag): lane<16 -> K = 0..15; lane>=16 -> K = 16..31
  int m = lane & 15;
  long row = (long)tile * 16 + m;
  int kbase = (lane < 16) ? 0 : 16;
  const float* srcs[3] = { X, Txo, Txi };
  v16h a[3];
#pragma unroll
  for (int s = 0; s < 3; ++s) {
    const float* p = srcs[s] + row * F_IN + kbase;
#pragma unroll
    for (int q = 0; q < 8; ++q) a[s][q] = (_Float16)p[q];
#pragma unroll
    for (int q = 0; q < 8; ++q) a[s][8 + q] = (_Float16)p[8 + q];
  }

#pragma unroll
  for (int nt = 0; nt < 8; ++nt) {
    v8f acc = {};
#pragma unroll
    for (int s = 0; s < 3; ++s) {
      v16h b = *(const v16h*)(sB + (((s * 8 + nt) * 32) + lane) * 16);
      acc = __builtin_amdgcn_wmma_f32_16x16x32_f16(
          /*neg_a=*/false, a[s], /*neg_b=*/false, b,
          /*c_mod=*/(short)0, acc, /*reuse_a=*/false, /*reuse_b=*/false);
    }
    // C/D 16x16 f32 layout: VGPR j -> M=j (lanes 0-15) / M=8+j (lanes 16-31); N = lane&15
    int colBase = nt * 16 + (lane & 15);
    long rlo = (long)tile * 16 + ((lane < 16) ? 0 : 8);
#pragma unroll
    for (int j = 0; j < 8; ++j)
      Pre[(rlo + j) * PRE_C + colBase] = acc[j];
  }
}

// ---------------- fused GRU nonlinearity + readout: one wave per node ----------------
__global__ void finalize_kernel(const float* __restrict__ Pre,
                                const float* __restrict__ bz, const float* __restrict__ bh,
                                const float* __restrict__ Wlin, const float* __restrict__ blin,
                                float* __restrict__ out, int n) {
  int wave = threadIdx.x >> 5, lane = threadIdx.x & 31;
  int node = blockIdx.x * (blockDim.x >> 5) + wave;
  if (node >= n) return;
  const float* p = Pre + (long)node * PRE_C;
  float acc = 0.0f;
#pragma unroll
  for (int h = 0; h < 2; ++h) {
    int f = lane + 32 * h;
    float z  = 1.0f / (1.0f + expf(-(p[f] + bz[f])));
    float ht = tanhf(p[64 + f] + bh[f]);
    float H  = (1.0f - z) * ht;                 // Z*H0 + (1-Z)*Ht with H0 = 0
    acc += fmaxf(H, 0.0f) * Wlin[f];            // relu(H) @ W_lin
  }
#pragma unroll
  for (int off = 16; off > 0; off >>= 1)
    acc += __shfl_down(acc, off, 32);
  if (lane == 0) out[node] = acc + blin[0];
}

extern "C" void kernel_launch(void* const* d_in, const int* in_sizes, int n_in,
                              void* d_out, int out_size, void* d_ws, size_t ws_size,
                              hipStream_t stream) {
  const float* x    = (const float*)d_in[0];
  const int*   ei   = (const int*)d_in[1];
  const float* w    = (const float*)d_in[2];
  const float* Wz   = (const float*)d_in[3];
  const float* bz   = (const float*)d_in[4];
  // d_in[5]=W_r, d_in[6]=b_r are dead (H0==0 -> H0*R==0)
  const float* Wh   = (const float*)d_in[7];
  const float* bh   = (const float*)d_in[8];
  const float* Wlin = (const float*)d_in[9];
  const float* blin = (const float*)d_in[10];
  float* out = (float*)d_out;

  const int N = in_sizes[0] / F_IN;
  const int E = in_sizes[2];
  const int* src = ei;
  const int* dst = ei + E;

  // workspace carve-up (256B aligned)
  char* ws = (char*)d_ws;
  size_t off = 0;
  auto carve = [&](size_t bytes) {
    char* p = ws + off;
    off += (bytes + 255) & ~((size_t)255);
    return p;
  };
  float*     degOut = (float*)carve((size_t)N * 4);
  float*     degIn  = (float*)carve((size_t)N * 4);
  float*     Txo    = (float*)carve((size_t)N * F_IN * 4);
  float*     Txi    = (float*)carve((size_t)N * F_IN * 4);
  _Float16*  Bfrag  = (_Float16*)carve((size_t)BFRAG_HALVES * 2);
  float*     Pre    = (float*)carve((size_t)N * PRE_C * 4);
  (void)ws_size;

  // 1) zero degrees + propagation accumulators (contiguous carve range)
  long nz = ((char*)Txi + (size_t)N * F_IN * 4 - (char*)degOut) / 4;
  zero_f32_kernel<<<1024, 256, 0, stream>>>(degOut, nz);

  // 2) weighted degrees, then in-place inversion
  degree_kernel<<<(E + 255) / 256, 256, 0, stream>>>(src, dst, w, degOut, degIn, E);
  invert_kernel<<<(N + 255) / 256, 256, 0, stream>>>(degOut, degIn, N);

  // 3) bidirectional diffusion propagation (shared by z and h gates)
  propagate_kernel<<<(E + 7) / 8, dim3(32, 8), 0, stream>>>(
      src, dst, w, x, degOut, degIn, Txo, Txi, E);

  // 4) pack effective weights into WMMA B-fragments (f16)
  pack_weights_kernel<<<(96 * PRE_C + 255) / 256, 256, 0, stream>>>(Wz, Wh, Bfrag);

  // 5) WMMA GEMM: Pre = [X|Txo|Txi] @ Wcat  (K = 96 = 3 x K32 steps), B via async LDS stage
  int nTiles = (N + 15) / 16;
  gemm_wmma_kernel<<<(nTiles + 7) / 8, 256, 0, stream>>>(x, Txo, Txi, Bfrag, Pre, nTiles);

  // 6) fused sigmoid/tanh GRU update + relu + linear readout
  finalize_kernel<<<(N + 7) / 8, 256, 0, stream>>>(Pre, bz, bh, Wlin, blin, out, N);
}
